// SimpleNeuralNets_82815559401717
// MI455X (gfx1250) — matrix operand order repaired and checked by reading, not compiled, not run on hardware
//
#include <hip/hip_runtime.h>

typedef unsigned int   u32;
typedef unsigned short u16;
typedef __attribute__((ext_vector_type(16))) __bf16 v16bf;
typedef __attribute__((ext_vector_type(8)))  float  v8f;

union BfFrag { v16bf v; uint4 q[2]; };

__device__ __forceinline__ u16 f2bf(float f) {
    u32 u = __float_as_uint(f);
    u += 0x7FFFu + ((u >> 16) & 1u);          // round-to-nearest-even
    return (u16)(u >> 16);
}

// ---------------------------------------------------------------- converters
__global__ void cvt_f32_to_bf16(const float* __restrict__ src,
                                u16* __restrict__ dst, int n) {
    int i = blockIdx.x * blockDim.x + threadIdx.x;
    if (i < n) dst[i] = f2bf(src[i]);
}

// ------------------------------------------------- gather + masked mean pool
// one 256-thread block per sample; D = 768 = 3 * 256
__global__ __launch_bounds__(256)
void gather_pool(const int* __restrict__ author_idx,
                 const int* __restrict__ section_idx,
                 const int* __restrict__ section_len,
                 const float* __restrict__ author_feat,
                 const float* __restrict__ section_feat,
                 u16* __restrict__ ae_bf,     // [B,768]
                 u16* __restrict__ comb_bf)   // [B,1536], cols 768..1535 written
{
    const int b = blockIdx.x;
    const int t = threadIdx.x;

    const int a = author_idx[b];
    const float* ap = author_feat + (size_t)a * 768;
    u16* aedst = ae_bf + (size_t)b * 768;
    aedst[t]       = f2bf(ap[t]);
    aedst[t + 256] = f2bf(ap[t + 256]);
    aedst[t + 512] = f2bf(ap[t + 512]);

    const int len = section_len[b];
    const int* sidx = section_idx + (size_t)b * 50;
    float acc0 = 0.f, acc1 = 0.f, acc2 = 0.f;
    for (int s = 0; s < len; ++s) {
        const float* sp = section_feat + (size_t)sidx[s] * 768;
        acc0 += sp[t];
        acc1 += sp[t + 256];
        acc2 += sp[t + 512];
    }
    const float inv = 1.0f / (float)(len < 1 ? 1 : len);
    u16* cdst = comb_bf + (size_t)b * 1536 + 768;
    cdst[t]       = f2bf(acc0 * inv);
    cdst[t + 256] = f2bf(acc1 * inv);
    cdst[t + 512] = f2bf(acc2 * inv);
}

// ------------------------------------------------------------- WMMA GEMM
// out[m,n] = act( sum_k A[m,k] * W[n,k] + bias[n] )
// A: [M, lda] bf16 row-major, W: [Ntiles*16, K] bf16 row-major.
// Block tile: 128 rows x 128 cols. 256 threads = 8 waves; each wave owns a
// 16-col N strip and 8 stacked 16x16 accumulators (B fragment reused 8x).
// A tile (128x32 bf16, 8KB) double-buffered in LDS, one barrier per K-step.
template <bool RELU, bool OUT_BF16>
__global__ __launch_bounds__(256)
void gemm_bf16_wmma(const u16* __restrict__ A, int lda,
                    const u16* __restrict__ W,
                    const float* __restrict__ bias,
                    void* __restrict__ outv, int ldo,
                    int Ntiles, int K)
{
    __shared__ __align__(16) u32 As[2][128 * 16];  // 2 x (128 rows x 32 bf16)

    const int lane   = threadIdx.x & 31;
    const int wave   = threadIdx.x >> 5;
    const int m0     = blockIdx.x << 7;            // 128 rows per block
    const int ntile  = (blockIdx.y << 3) + wave;
    const bool active = ntile < Ntiles;

    // cooperative A staging: two 16B chunks per thread (128 rows x 64B/row)
    const int arow = threadIdx.x >> 1;             // 0..127
    const int aseg = threadIdx.x & 1;              // 0..1 (16 bf16 each)
    const u16* aptr = A + (size_t)(m0 + arow) * lda + (aseg << 4);
    const int asl = (arow << 4) + (aseg << 3);     // u32 slot in LDS tile

    const int mcol = lane & 15;                    // row/col offset in tile
    const int half = lane >> 4;                    // 0 or 1 (lane half)
    const u16* wrow = W + (size_t)((active ? ntile : 0) * 16 + mcol) * K;

    v8f c[8] = {};

    // prologue: stage K-slab 0 into buffer 0
    uint4 st0 = *(const uint4*)(aptr);
    uint4 st1 = *(const uint4*)(aptr + 8);
    *(uint4*)&As[0][asl]     = st0;
    *(uint4*)&As[0][asl + 4] = st1;

    int cur = 0;
    for (int k0 = 0; k0 < K; k0 += 32) {
        __syncthreads();                           // As[cur] ready
        const bool more = (k0 + 32) < K;
        if (more) {                                // next slab -> regs
            st0 = *(const uint4*)(aptr + k0 + 32);
            st1 = *(const uint4*)(aptr + k0 + 40);
        }

        if (active) {
            // B frag: lane holds column n; lanes 0-15 -> K{0..7,16..23}, 16-31 -> K{8..15,24..31}
            BfFrag bm;
            const u16* wp = wrow + k0 + (half << 3);
            bm.q[0] = *(const uint4*)(wp);
            bm.q[1] = *(const uint4*)(wp + 16);
            __builtin_prefetch(wp + 32, 0, 3);     // next K-step of this B row

            // load ALL A fragments first so the ds_loads overlap, then issue
            // the 8 WMMAs back-to-back (independent accumulators, no hazards)
            BfFrag af[8];
            #pragma unroll
            for (int ms = 0; ms < 8; ++ms) {
                // A frag (ISA 16-bit A layout): lanes 0-15 -> K{0..7,16..23},
                //                               lanes 16-31 -> K{8..15,24..31}
                const u32* asb = &As[cur][((ms << 4) + mcol) << 4];
                af[ms].q[0] = *(const uint4*)&asb[(half << 2)];
                af[ms].q[1] = *(const uint4*)&asb[8 + (half << 2)];
            }
            #pragma unroll
            for (int ms = 0; ms < 8; ++ms) {
                c[ms] = __builtin_amdgcn_wmma_f32_16x16x32_bf16(
                            false, af[ms].v, false, bm.v, (short)0, c[ms],
                            false, false);
            }
        }

        if (more) {                                // fill other buffer
            *(uint4*)&As[cur ^ 1][asl]     = st0;
            *(uint4*)&As[cur ^ 1][asl + 4] = st1;
        }
        cur ^= 1;
    }

    if (active) {
        const int n    = ntile * 16 + mcol;
        const float bv = bias[n];
        #pragma unroll
        for (int ms = 0; ms < 8; ++ms) {
            #pragma unroll
            for (int r = 0; r < 8; ++r) {
                float v = c[ms][r] + bv;
                if (RELU) v = fmaxf(v, 0.0f);
                const size_t o =
                    (size_t)(m0 + (ms << 4) + r + (half << 3)) * ldo + n;
                if (OUT_BF16) ((u16*)outv)[o]   = f2bf(v);
                else          ((float*)outv)[o] = v;
            }
        }
    }
}

// ---------------------------------------------------------------- launcher
extern "C" void kernel_launch(void* const* d_in, const int* in_sizes, int n_in,
                              void* d_out, int out_size, void* d_ws, size_t ws_size,
                              hipStream_t stream)
{
    (void)in_sizes; (void)n_in; (void)out_size; (void)ws_size;

    const int*   author_idx   = (const int*)  d_in[0];
    const int*   section_idx  = (const int*)  d_in[1];
    const int*   section_len  = (const int*)  d_in[2];
    const float* author_feat  = (const float*)d_in[3];
    const float* section_feat = (const float*)d_in[4];
    const float* Wa = (const float*)d_in[5];
    const float* ba = (const float*)d_in[6];
    const float* Wc = (const float*)d_in[7];
    const float* bc = (const float*)d_in[8];
    const float* Wo = (const float*)d_in[9];
    const float* bo = (const float*)d_in[10];
    float* out = (float*)d_out;

    const int B = 4096, D = 768, C = 10000;

    char* ws = (char*)d_ws;
    size_t off = 0;
    auto salloc = [&](size_t bytes) -> void* {
        off = (off + 255) & ~(size_t)255;
        void* p = ws + off;
        off += bytes;
        return p;
    };
    u16* Wa_bf   = (u16*)salloc((size_t)D * D * 2);
    u16* Wc_bf   = (u16*)salloc((size_t)D * 2 * D * 2);
    u16* Wo_bf   = (u16*)salloc((size_t)C * D * 2);
    u16* ae_bf   = (u16*)salloc((size_t)B * D * 2);
    u16* comb_bf = (u16*)salloc((size_t)B * 2 * D * 2);
    u16* h_bf    = (u16*)salloc((size_t)B * D * 2);

    // weights -> bf16
    cvt_f32_to_bf16<<<(D * D + 255) / 256, 256, 0, stream>>>(Wa, Wa_bf, D * D);
    cvt_f32_to_bf16<<<(D * 2 * D + 255) / 256, 256, 0, stream>>>(Wc, Wc_bf, D * 2 * D);
    cvt_f32_to_bf16<<<(C * D + 255) / 256, 256, 0, stream>>>(Wo, Wo_bf, C * D);

    // gathers + mean pool (fills ae_bf and comb_bf[:, D:2D])
    gather_pool<<<B, 256, 0, stream>>>(author_idx, section_idx, section_len,
                                       author_feat, section_feat, ae_bf, comb_bf);

    // GEMM1: x_author = relu(ae @ Wa^T + ba) -> comb_bf[:, 0:D]
    dim3 gAB(B / 128, (D / 16 + 7) / 8);           // (32, 6)
    gemm_bf16_wmma<true, true><<<gAB, 256, 0, stream>>>(
        ae_bf, D, Wa_bf, ba, (void*)comb_bf, 2 * D, D / 16, D);

    // GEMM2: h = comb @ Wc^T + bc  (no relu) -> h_bf
    gemm_bf16_wmma<false, true><<<gAB, 256, 0, stream>>>(
        comb_bf, 2 * D, Wc_bf, bc, (void*)h_bf, D, D / 16, 2 * D);

    // GEMM3: out = relu(h @ Wo^T + bo) -> f32 logits
    dim3 gO(B / 128, (C / 16 + 7) / 8);            // (32, 79)
    gemm_bf16_wmma<true, false><<<gO, 256, 0, stream>>>(
        h_bf, D, Wo_bf, bo, (void*)out, C, C / 16, D);
}
